// CrossAttention_72739566125350
// MI455X (gfx1250) — matrix operand order repaired and checked by reading, not compiled, not run on hardware
//
#include <hip/hip_runtime.h>

// ---------------------------------------------------------------------------
// CDNA5 (gfx1250) cross-attention: bf16 WMMA GEMMs + flash attention.
// B=8, S=4096, D_IN=1024, L=64, D_LAT=512, H=16, DH=64
// ---------------------------------------------------------------------------

typedef __bf16 bf16;
typedef __bf16 v16bf __attribute__((ext_vector_type(16)));
typedef __bf16 v8bf  __attribute__((ext_vector_type(8)));
typedef float  v8f   __attribute__((ext_vector_type(8)));

#define CA_B    8
#define CA_S    4096
#define CA_DIN  1024
#define CA_L    64
#define CA_DLAT 512
#define CA_H    16
#define CA_DH   64
#define CA_N    1024   // QK_DIM == V_DIM == D_OUT

union Frag  { v16bf v; v8bf h[2]; };
union Pack8 { v8bf v; bf16 e[8]; };
union Pack2 { unsigned u; bf16 e[2]; };

__device__ inline v8f v8f_zero() {
    v8f r;
#pragma unroll
    for (int i = 0; i < 8; ++i) r[i] = 0.0f;
    return r;
}

// ---------------------------------------------------------------------------
// Generic C = A(fp32)[M,K] * B(fp32)[K,N] + bias, output bf16 or f32.
// Block tile 128(M) x 128(N), K-step 32. 256 threads = 8 waves (4 M x 2 N),
// each wave computes 32x64 via 2x4 grid of 16x16x32 bf16 WMMA accumulators.
// Software-pipelined: global loads for tile k+32 overlap WMMAs of tile k.
// ---------------------------------------------------------------------------
template <bool OUT_BF16>
__global__ __launch_bounds__(256) void wmma_gemm(
    const float* __restrict__ A, const float* __restrict__ Bm,
    const float* __restrict__ bias, void* __restrict__ Cout,
    int M, int N, int K)
{
    __shared__ __align__(16) bf16 As[128][48];   // [m][k], padded
    __shared__ __align__(16) bf16 Bt[128][48];   // [n][k] (transposed), padded

    const int tid   = threadIdx.x;
    const int wave  = tid >> 5;
    const int lane  = tid & 31;
    const int lan16 = lane & 15;
    const int g     = lane >> 4;       // 0 or 1 (lane half)
    const int wm    = wave & 3;        // M sub-tile: wm*32
    const int wn    = wave >> 2;       // N sub-tile: wn*64
    const int mBase = blockIdx.y * 128;
    const int nBase = blockIdx.x * 128;

    // staging thread mappings
    const int arow  = tid >> 1;         // 0..127
    const int akoff = (tid & 1) * 16;   // 0 or 16
    const int bkr   = (tid >> 4) * 2;   // 0,2,..,30  (two k-rows per thread)
    const int bnoff = (tid & 15) * 8;   // 0..120     (8 n per thread)

    v8f acc[2][4];
#pragma unroll
    for (int mi = 0; mi < 2; ++mi)
#pragma unroll
        for (int ni = 0; ni < 4; ++ni) acc[mi][ni] = v8f_zero();

    float aReg[16], b0Reg[8], b1Reg[8];
    const bool rok = (mBase + arow) < M;

    auto gload = [&](int k0) {
        const float* asrc = A + (size_t)(mBase + arow) * K + k0 + akoff;
#pragma unroll
        for (int i = 0; i < 16; ++i) aReg[i] = rok ? asrc[i] : 0.0f;
        const float* bsrc = Bm + (size_t)(k0 + bkr) * N + nBase + bnoff;
#pragma unroll
        for (int i = 0; i < 8; ++i) b0Reg[i] = bsrc[i];
#pragma unroll
        for (int i = 0; i < 8; ++i) b1Reg[i] = bsrc[N + i];
    };

    gload(0);

    for (int k0 = 0; k0 < K; k0 += 32) {
        // --- convert + vectorized store of staged tile to LDS -------------
#pragma unroll
        for (int half = 0; half < 2; ++half) {
            Pack8 pa;
#pragma unroll
            for (int i = 0; i < 8; ++i) pa.e[i] = (bf16)aReg[half * 8 + i];
            *(v8bf*)&As[arow][akoff + half * 8] = pa.v;   // ds_store_b128
        }
#pragma unroll
        for (int i = 0; i < 8; ++i) {
            Pack2 pb;
            pb.e[0] = (bf16)b0Reg[i];
            pb.e[1] = (bf16)b1Reg[i];
            *(unsigned*)&Bt[bnoff + i][bkr] = pb.u;       // ds_store_b32
        }
        __syncthreads();

        // --- prefetch next tile into registers (overlaps WMMAs below) ----
        if (k0 + 32 < K) gload(k0 + 32);

        // --- fragments ----------------------------------------------------
        Frag af[2], bfr[4];
#pragma unroll
        for (int mi = 0; mi < 2; ++mi) {
            const int row = wm * 32 + mi * 16 + lan16;
            af[mi].h[0] = *(const v8bf*)&As[row][g * 8];
            af[mi].h[1] = *(const v8bf*)&As[row][16 + g * 8];
        }
#pragma unroll
        for (int ni = 0; ni < 4; ++ni) {
            const int col = wn * 64 + ni * 16 + lan16;
            bfr[ni].h[0] = *(const v8bf*)&Bt[col][g * 8];
            bfr[ni].h[1] = *(const v8bf*)&Bt[col][16 + g * 8];
        }
#pragma unroll
        for (int mi = 0; mi < 2; ++mi)
#pragma unroll
            for (int ni = 0; ni < 4; ++ni)
                acc[mi][ni] = __builtin_amdgcn_wmma_f32_16x16x32_bf16(
                    false, af[mi].v, false, bfr[ni].v,
                    (short)0, acc[mi][ni], false, false);
        __syncthreads();
    }

    // --- epilogue: bias + store (C layout: VGPR j -> M=j+8*g, N=lan16) ----
#pragma unroll
    for (int mi = 0; mi < 2; ++mi) {
#pragma unroll
        for (int ni = 0; ni < 4; ++ni) {
            const int col = nBase + wn * 64 + ni * 16 + lan16;
            const float bv = bias[col];
#pragma unroll
            for (int j = 0; j < 8; ++j) {
                const int row = mBase + wm * 32 + mi * 16 + j + 8 * g;
                if (row < M) {
                    const float v = acc[mi][ni][j] + bv;
                    if constexpr (OUT_BF16)
                        ((bf16*)Cout)[(size_t)row * N + col] = (bf16)v;
                    else
                        ((float*)Cout)[(size_t)row * N + col] = v;
                }
            }
        }
    }
}

// ---------------------------------------------------------------------------
// Flash attention: one block per (b,h). 4 waves, wave w owns L rows
// [w*16, w*16+16). Streams S in 64-key blocks with online softmax.
// K block staged with CDNA5 async global->LDS copies (ASYNCcnt).
// Q: bf16 [L, 1024] (shared across batch); K,V: bf16 [B, S, H, DH]
// Out: f32 [B, L, H*DH]
// ---------------------------------------------------------------------------
__global__ __launch_bounds__(128) void attn_kernel(
    const bf16* __restrict__ Q, const bf16* __restrict__ Kb,
    const bf16* __restrict__ Vb, float* __restrict__ Out)
{
    __shared__ __align__(16) bf16 Ks[64][72];       // [s][d]
    __shared__ __align__(16) bf16 Vt[64][72];       // [d][s] (transposed)
    __shared__ __align__(16) bf16 Ps[4][16][72];    // per-wave P staging

    const int bh    = blockIdx.x;
    const int b     = bh / CA_H;
    const int h     = bh % CA_H;
    const int tid   = threadIdx.x;
    const int wave  = tid >> 5;
    const int lane  = tid & 31;
    const int lan16 = lane & 15;
    const int g     = lane >> 4;

    // LDS byte offset of Ks (generic LDS pointer truncates to LDS address)
    const unsigned ksLds = (unsigned)(uintptr_t)(void*)&Ks[0][0];

    // persistent Q fragments (A layout), rows w*16+lan16, d split in 2x32
    Frag aq[2];
    {
        const bf16* qrow = Q + (size_t)(wave * 16 + lan16) * CA_N + h * CA_DH;
#pragma unroll
        for (int c = 0; c < 2; ++c) {
            aq[c].h[0] = *(const v8bf*)(qrow + c * 32 + g * 8);
            aq[c].h[1] = *(const v8bf*)(qrow + c * 32 + 16 + g * 8);
        }
    }

    float m[8], lsum[8];
    v8f o[4];
#pragma unroll
    for (int j = 0; j < 8; ++j) { m[j] = -3.0e38f; lsum[j] = 0.0f; }
#pragma unroll
    for (int nf = 0; nf < 4; ++nf) o[nf] = v8f_zero();

    const float scale = 0.125f;   // DH^-0.5

    for (int sb = 0; sb < CA_S / 64; ++sb) {
        __syncthreads();   // protect Ks/Vt against previous iteration readers

        // ---- K block: async global->LDS copy (64 rows x 128B) ------------
        {
            const size_t gbase = (((size_t)b * CA_S + sb * 64) * CA_H + h) * CA_DH;
#pragma unroll
            for (int it = 0; it < 4; ++it) {
                const int c    = tid + it * 128;        // 0..511 16B chunks
                const int srow = c >> 3;
                const int o16  = c & 7;
                const unsigned dst = ksLds + (unsigned)(srow * 144 + o16 * 16);
                const bf16* src = Kb + gbase + (size_t)srow * (CA_H * CA_DH) + o16 * 8;
                asm volatile("global_load_async_to_lds_b128 %0, %1, off"
                             :: "v"(dst), "v"(src)
                             : "memory");
            }
        }
        // ---- V block: manual transpose, packed b32 LDS writes ------------
        {
            const int sr   = (lane) * 2 + 0;            // two s rows / thread
            const int srr  = (tid & 31) * 2;
            const int doff = (tid >> 5) * 16;
            (void)sr;
            const size_t vb = (((size_t)b * CA_S + sb * 64 + srr) * CA_H + h) * CA_DH + doff;
            Pack8 r0a, r0b, r1a, r1b;
            r0a.v = *(const v8bf*)(Vb + vb);
            r0b.v = *(const v8bf*)(Vb + vb + 8);
            r1a.v = *(const v8bf*)(Vb + vb + CA_H * CA_DH);
            r1b.v = *(const v8bf*)(Vb + vb + CA_H * CA_DH + 8);
#pragma unroll
            for (int i = 0; i < 16; ++i) {
                Pack2 p;
                p.e[0] = (i < 8) ? r0a.e[i] : r0b.e[i - 8];
                p.e[1] = (i < 8) ? r1a.e[i] : r1b.e[i - 8];
                *(unsigned*)&Vt[doff + i][srr] = p.u;    // ds_store_b32
            }
        }
        asm volatile("s_wait_asynccnt 0" ::: "memory");
        __syncthreads();

        // scores S = (Q Kt) * scale : 16 x 64 per wave
        v8f sc[4];
#pragma unroll
        for (int nf = 0; nf < 4; ++nf) {
            v8f a = v8f_zero();
#pragma unroll
            for (int c = 0; c < 2; ++c) {
                Frag bk;
                const bf16* p = &Ks[nf * 16 + lan16][c * 32 + g * 8];
                bk.h[0] = *(const v8bf*)p;
                bk.h[1] = *(const v8bf*)(p + 16);
                a = __builtin_amdgcn_wmma_f32_16x16x32_bf16(
                    false, aq[c].v, false, bk.v, (short)0, a, false, false);
            }
#pragma unroll
            for (int j = 0; j < 8; ++j) sc[nf][j] = a[j] * scale;
        }

        // row max across the 64 keys (4 frags elementwise + xor-shuffle 16)
        float alpha[8];
#pragma unroll
        for (int j = 0; j < 8; ++j) {
            float v = fmaxf(fmaxf(sc[0][j], sc[1][j]), fmaxf(sc[2][j], sc[3][j]));
            v = fmaxf(v, __shfl_xor(v, 1));
            v = fmaxf(v, __shfl_xor(v, 2));
            v = fmaxf(v, __shfl_xor(v, 4));
            v = fmaxf(v, __shfl_xor(v, 8));
            const float mn = fmaxf(m[j], v);
            alpha[j] = __expf(m[j] - mn);
            m[j] = mn;
        }

        // P = exp(S - m), row sums
        float rs[8];
#pragma unroll
        for (int j = 0; j < 8; ++j) rs[j] = 0.0f;
#pragma unroll
        for (int nf = 0; nf < 4; ++nf)
#pragma unroll
            for (int j = 0; j < 8; ++j) {
                const float p = __expf(sc[nf][j] - m[j]);
                sc[nf][j] = p;
                rs[j] += p;
            }
#pragma unroll
        for (int j = 0; j < 8; ++j) {
            float v = rs[j];
            v += __shfl_xor(v, 1);
            v += __shfl_xor(v, 2);
            v += __shfl_xor(v, 4);
            v += __shfl_xor(v, 8);
            lsum[j] = lsum[j] * alpha[j] + v;
        }
#pragma unroll
        for (int nf = 0; nf < 4; ++nf)
#pragma unroll
            for (int j = 0; j < 8; ++j) o[nf][j] *= alpha[j];

        // stage P (C layout -> LDS -> A layout)
#pragma unroll
        for (int nf = 0; nf < 4; ++nf)
#pragma unroll
            for (int j = 0; j < 8; ++j)
                Ps[wave][j + 8 * g][nf * 16 + lan16] = (bf16)sc[nf][j];
        __syncthreads();

        Frag ap[2];
#pragma unroll
        for (int c = 0; c < 2; ++c) {
            ap[c].h[0] = *(const v8bf*)&Ps[wave][lan16][c * 32 + g * 8];
            ap[c].h[1] = *(const v8bf*)&Ps[wave][lan16][c * 32 + 16 + g * 8];
        }

        // O += P @ V
#pragma unroll
        for (int nf = 0; nf < 4; ++nf) {
#pragma unroll
            for (int c = 0; c < 2; ++c) {
                Frag bv;
                const bf16* p = &Vt[nf * 16 + lan16][c * 32 + g * 8];
                bv.h[0] = *(const v8bf*)p;
                bv.h[1] = *(const v8bf*)(p + 16);
                o[nf] = __builtin_amdgcn_wmma_f32_16x16x32_bf16(
                    false, ap[c].v, false, bv.v, (short)0, o[nf], false, false);
            }
        }
    }

    // normalize and store: Out[b, l, h*64 + d]
#pragma unroll
    for (int j = 0; j < 8; ++j) lsum[j] = 1.0f / lsum[j];
#pragma unroll
    for (int nf = 0; nf < 4; ++nf)
#pragma unroll
        for (int j = 0; j < 8; ++j) {
            const int l = wave * 16 + j + 8 * g;
            Out[((size_t)b * CA_L + l) * CA_N + h * CA_DH + nf * 16 + lan16] =
                o[nf][j] * lsum[j];
        }
}

// ---------------------------------------------------------------------------
// Final: out = (LayerNorm(y) + res[l]) * rsqrt(2). One block per (b,l) row.
// ---------------------------------------------------------------------------
__global__ __launch_bounds__(256) void ln_res_kernel(
    const float* __restrict__ Y, const float* __restrict__ Res,
    const float* __restrict__ gamma, const float* __restrict__ beta,
    float* __restrict__ Outp)
{
    __shared__ float red[256];
    __shared__ float red2[256];
    const int row = blockIdx.x;        // b*L + l
    const int l = row % CA_L;
    const float* y = Y + (size_t)row * CA_N;

    float s = 0.0f, s2 = 0.0f;
    float vals[4];
#pragma unroll
    for (int i = 0; i < 4; ++i) {
        const float v = y[threadIdx.x + i * 256];
        vals[i] = v; s += v; s2 += v * v;
    }
    red[threadIdx.x] = s;
    red2[threadIdx.x] = s2;
    __syncthreads();
    for (int off = 128; off > 0; off >>= 1) {
        if (threadIdx.x < off) {
            red[threadIdx.x]  += red[threadIdx.x + off];
            red2[threadIdx.x] += red2[threadIdx.x + off];
        }
        __syncthreads();
    }
    const float mu   = red[0] * (1.0f / CA_N);
    const float var  = red2[0] * (1.0f / CA_N) - mu * mu;
    const float rstd = rsqrtf(var + 1e-5f);
    const float rs2  = 0.70710678118654752440f;
#pragma unroll
    for (int i = 0; i < 4; ++i) {
        const int c = threadIdx.x + i * 256;
        const float v = (vals[i] - mu) * rstd * gamma[c] + beta[c]
                      + Res[(size_t)l * CA_N + c];
        Outp[(size_t)row * CA_N + c] = v * rs2;
    }
}

// ---------------------------------------------------------------------------
static inline size_t alignup(size_t x) { return (x + 255) & ~(size_t)255; }

extern "C" void kernel_launch(void* const* d_in, const int* in_sizes, int n_in,
                              void* d_out, int out_size, void* d_ws, size_t ws_size,
                              hipStream_t stream)
{
    const float* x    = (const float*)d_in[0];
    const float* lat  = (const float*)d_in[1];
    const float* Wq   = (const float*)d_in[2];
    const float* bq   = (const float*)d_in[3];
    const float* Wk   = (const float*)d_in[4];
    const float* bk   = (const float*)d_in[5];
    const float* Wv   = (const float*)d_in[6];
    const float* bv   = (const float*)d_in[7];
    const float* Wo   = (const float*)d_in[8];
    const float* bo   = (const float*)d_in[9];
    const float* Wres = (const float*)d_in[10];
    const float* bres = (const float*)d_in[11];
    const float* ln_g = (const float*)d_in[12];
    const float* ln_b = (const float*)d_in[13];
    (void)in_sizes; (void)n_in; (void)out_size; (void)ws_size;

    char* ws = (char*)d_ws;
    bf16* qb = (bf16*)ws;  ws += alignup((size_t)CA_L * CA_N * sizeof(bf16));
    bf16* Kb = (bf16*)ws;  ws += alignup((size_t)CA_B * CA_S * CA_N * sizeof(bf16));
    bf16* Vb = (bf16*)ws;  ws += alignup((size_t)CA_B * CA_S * CA_N * sizeof(bf16));
    float* ao  = (float*)ws; ws += alignup((size_t)CA_B * CA_L * CA_N * sizeof(float));
    float* yt  = (float*)ws; ws += alignup((size_t)CA_B * CA_L * CA_N * sizeof(float));
    float* res = (float*)ws; ws += alignup((size_t)CA_L * CA_N * sizeof(float));

    const dim3 blk(256);
    // q = latents @ Wq + bq            [64, 1024], K=512   -> bf16
    wmma_gemm<true><<<dim3(8, 1), blk, 0, stream>>>(lat, Wq, bq, qb, CA_L, CA_N, CA_DLAT);
    // K = x @ Wk + bk                  [32768, 1024]       -> bf16
    wmma_gemm<true><<<dim3(8, 256), blk, 0, stream>>>(x, Wk, bk, Kb, CA_B * CA_S, CA_N, CA_DIN);
    // V = x @ Wv + bv                  [32768, 1024]       -> bf16
    wmma_gemm<true><<<dim3(8, 256), blk, 0, stream>>>(x, Wv, bv, Vb, CA_B * CA_S, CA_N, CA_DIN);
    // attention                        -> ao f32 [B, L, 1024]
    attn_kernel<<<dim3(CA_B * CA_H), dim3(128), 0, stream>>>(qb, Kb, Vb, ao);
    // y = ao @ Wo + bo                 [512, 1024]         -> f32
    wmma_gemm<false><<<dim3(8, 4), blk, 0, stream>>>(ao, Wo, bo, yt, CA_B * CA_L, CA_N, CA_N);
    // res = latents @ Wres + bres      [64, 1024], K=512   -> f32
    wmma_gemm<false><<<dim3(8, 1), blk, 0, stream>>>(lat, Wres, bres, res, CA_L, CA_N, CA_DLAT);
    // final layernorm + residual + rsqrt(2)
    ln_res_kernel<<<dim3(CA_B * CA_L), dim3(256), 0, stream>>>(yt, res, ln_g, ln_b, (float*)d_out);
}
